// GrahamLoss_50878182588704
// MI455X (gfx1250) — matrix operand order repaired and checked by reading, not compiled, not run on hardware
//
#include <hip/hip_runtime.h>

// ---- CDNA5 WMMA types ----
typedef __attribute__((ext_vector_type(16))) __bf16       v16bf;
typedef __attribute__((ext_vector_type(2)))  __bf16       v2bf;
typedef __attribute__((ext_vector_type(2)))  float        v2f;
typedef __attribute__((ext_vector_type(8)))  float        v8f;
typedef __attribute__((ext_vector_type(4)))  unsigned int v4u;

#define C_DIM      512
#define HW_DIM     1024
#define KSTEP      32
#define TILE       64
#define LDS_STRIDE 40   // bf16 elems per row: 32 used + 8 pad (80B rows, 16B aligned chunks)
#define NTILEPAIR  36   // 8*9/2 tile pairs (tm <= tn) over 8x8 tiles of C=512
#define NBLOCKS    (16 * NTILEPAIR)

union FragU {
    v16bf bf;
    v4u   u[2];
};

// Convert two f32 -> packed bf16 pair (lo in [15:0], hi in [31:16]).
// Expressed as a 2-wide vector fptrunc so the backend can match
// v_cvt_pk_bf16_f32 on gfx1250.
__device__ __forceinline__ unsigned int cvt2bf(float lo, float hi) {
    v2f t;
    t.x = lo;
    t.y = hi;
    v2bf pk = __builtin_convertvector(t, v2bf);
    return __builtin_bit_cast(unsigned int, pk);
}

__device__ __forceinline__ v4u pack8(float4 a, float4 b) {
    v4u r;
    r.x = cvt2bf(a.x, a.y);
    r.y = cvt2bf(a.z, a.w);
    r.z = cvt2bf(b.x, b.y);
    r.w = cvt2bf(b.z, b.w);
    return r;
}

__global__ void __launch_bounds__(256)
gram_diff_tile_kernel(const float* __restrict__ fe,
                      const float* __restrict__ fd,
                      float* __restrict__ partial) {
    __shared__ unsigned short lAe[TILE * LDS_STRIDE];
    __shared__ unsigned short lAd[TILE * LDS_STRIDE];
    __shared__ unsigned short lBe[TILE * LDS_STRIDE];
    __shared__ unsigned short lBd[TILE * LDS_STRIDE];
    __shared__ float red[8];

    const int blk = blockIdx.x;
    const int b   = blk / NTILEPAIR;
    int p = blk % NTILEPAIR;

    // triangular unrank: (tm, tn) with tm <= tn over 8x8 tiles
    int tm = 0;
    while (p >= (8 - tm)) { p -= (8 - tm); ++tm; }
    const int tn = tm + p;

    const float* baseE = fe + (size_t)b * C_DIM * HW_DIM;
    const float* baseD = fd + (size_t)b * C_DIM * HW_DIM;
    const int rowA0 = tm * TILE;   // m rows
    const int rowB0 = tn * TILE;   // n rows

    const int tid  = threadIdx.x;
    const int lane = tid & 31;
    const int wave = tid >> 5;
    const int wr   = wave >> 1;          // wave's 16-row sub-tile (0..3)
    const int wc   = (wave & 1) * 2;     // first of two 16-col sub-tiles (0 or 2)

    v8f accE0 = {}, accE1 = {}, accD0 = {}, accD1 = {};

    // global->LDS staging: each thread loads 8 floats (2 float4) per buffer
    const int ldr = tid >> 2;            // row within 64-row tile
    const int ldk = (tid & 3) * 8;       // k offset: 0,8,16,24

    const float4* pAe = (const float4*)(baseE + (size_t)(rowA0 + ldr) * HW_DIM + ldk);
    const float4* pAd = (const float4*)(baseD + (size_t)(rowA0 + ldr) * HW_DIM + ldk);
    const float4* pBe = (const float4*)(baseE + (size_t)(rowB0 + ldr) * HW_DIM + ldk);
    const float4* pBd = (const float4*)(baseD + (size_t)(rowB0 + ldr) * HW_DIM + ldk);

    // precomputed LDS addresses
    v4u* const stAe = (v4u*)&lAe[ldr * LDS_STRIDE + ldk];
    v4u* const stAd = (v4u*)&lAd[ldr * LDS_STRIDE + ldk];
    v4u* const stBe = (v4u*)&lBe[ldr * LDS_STRIDE + ldk];
    v4u* const stBd = (v4u*)&lBd[ldr * LDS_STRIDE + ldk];

    const int h     = lane >> 4;          // lane half
    const int mrow  = wr * 16 + (lane & 15);
    const int nrow0 = wc * 16 + (lane & 15);
    const int nrow1 = nrow0 + 16;

    for (int kk = 0; kk < HW_DIM; kk += KSTEP) {
        __syncthreads();

        // ---- stage 64x32 f32 -> bf16 LDS for all four operand blocks ----
        {
            float4 ae0 = pAe[0], ae1 = pAe[1];
            float4 ad0 = pAd[0], ad1 = pAd[1];
            float4 be0 = pBe[0], be1 = pBe[1];
            float4 bd0 = pBd[0], bd1 = pBd[1];
            *stAe = pack8(ae0, ae1);
            *stAd = pack8(ad0, ad1);
            *stBe = pack8(be0, be1);
            *stBd = pack8(bd0, bd1);
        }

        // prefetch next k-step lines into cache (gfx1250 global_prefetch)
        if (kk + KSTEP < HW_DIM) {
            __builtin_prefetch(pAe + 8, 0, 3);
            __builtin_prefetch(pAd + 8, 0, 3);
            __builtin_prefetch(pBe + 8, 0, 3);
            __builtin_prefetch(pBd + 8, 0, 3);
        }
        pAe += 8; pAd += 8; pBe += 8; pBd += 8;   // advance 32 floats

        __syncthreads();

        // A fragments (16x32 bf16): lane half h holds K in {16h..16h+15} contiguous
        FragU aE, aD;
        aE.u[0] = *(const v4u*)&lAe[mrow * LDS_STRIDE + 16 * h];
        aE.u[1] = *(const v4u*)&lAe[mrow * LDS_STRIDE + 16 * h + 8];
        aD.u[0] = *(const v4u*)&lAd[mrow * LDS_STRIDE + 16 * h];
        aD.u[1] = *(const v4u*)&lAd[mrow * LDS_STRIDE + 16 * h + 8];

        // B fragments (32x16 bf16): lane half h holds K in {16h..16h+15} contiguous
        FragU bE0, bE1, bD0, bD1;
        bE0.u[0] = *(const v4u*)&lBe[nrow0 * LDS_STRIDE + 16 * h];
        bE0.u[1] = *(const v4u*)&lBe[nrow0 * LDS_STRIDE + 16 * h + 8];
        bE1.u[0] = *(const v4u*)&lBe[nrow1 * LDS_STRIDE + 16 * h];
        bE1.u[1] = *(const v4u*)&lBe[nrow1 * LDS_STRIDE + 16 * h + 8];
        bD0.u[0] = *(const v4u*)&lBd[nrow0 * LDS_STRIDE + 16 * h];
        bD0.u[1] = *(const v4u*)&lBd[nrow0 * LDS_STRIDE + 16 * h + 8];
        bD1.u[0] = *(const v4u*)&lBd[nrow1 * LDS_STRIDE + 16 * h];
        bD1.u[1] = *(const v4u*)&lBd[nrow1 * LDS_STRIDE + 16 * h + 8];

        // ---- 4 WMMAs per k-step per wave ----
        accE0 = __builtin_amdgcn_wmma_f32_16x16x32_bf16(false, aE.bf, false, bE0.bf,
                                                        (short)0, accE0, false, false);
        accE1 = __builtin_amdgcn_wmma_f32_16x16x32_bf16(false, aE.bf, false, bE1.bf,
                                                        (short)0, accE1, false, false);
        accD0 = __builtin_amdgcn_wmma_f32_16x16x32_bf16(false, aD.bf, false, bD0.bf,
                                                        (short)0, accD0, false, false);
        accD1 = __builtin_amdgcn_wmma_f32_16x16x32_bf16(false, aD.bf, false, bD1.bf,
                                                        (short)0, accD1, false, false);
    }

    // ---- epilogue: squared diff, weighted by tile symmetry ----
    const float w = (tm == tn) ? 1.0f : 2.0f;
    float s = 0.0f;
#pragma unroll
    for (int i = 0; i < 8; ++i) {
        float d0 = accD0[i] - accE0[i];
        float d1 = accD1[i] - accE1[i];
        s += d0 * d0 + d1 * d1;
    }
    s *= w;

    // wave32 reduction
#pragma unroll
    for (int off = 16; off > 0; off >>= 1)
        s += __shfl_xor(s, off, 32);
    if (lane == 0) red[wave] = s;
    __syncthreads();
    if (tid == 0) {
        float t = 0.0f;
#pragma unroll
        for (int i = 0; i < 8; ++i) t += red[i];
        partial[blk] = t;
    }
}

__global__ void __launch_bounds__(256)
final_reduce_kernel(const float* __restrict__ partial, float* __restrict__ out,
                    int n, float scale) {
    __shared__ float red[8];
    float s = 0.0f;
    for (int i = threadIdx.x; i < n; i += 256) s += partial[i];
#pragma unroll
    for (int off = 16; off > 0; off >>= 1)
        s += __shfl_xor(s, off, 32);
    const int lane = threadIdx.x & 31, wave = threadIdx.x >> 5;
    if (lane == 0) red[wave] = s;
    __syncthreads();
    if (threadIdx.x == 0) {
        float t = 0.0f;
#pragma unroll
        for (int i = 0; i < 8; ++i) t += red[i];
        out[0] = t * scale;
    }
}

extern "C" void kernel_launch(void* const* d_in, const int* in_sizes, int n_in,
                              void* d_out, int out_size, void* d_ws, size_t ws_size,
                              hipStream_t stream) {
    const float* feat       = (const float*)d_in[0];
    const float* feat_decod = (const float*)d_in[1];
    float* out = (float*)d_out;
    float* ws  = (float*)d_ws;

    gram_diff_tile_kernel<<<NBLOCKS, 256, 0, stream>>>(feat, feat_decod, ws);

    // denom = 4 * hw^2 * C^2 = 2^40
    const float scale = 1.0f / (4.0f * 1024.0f * 1024.0f * 512.0f * 512.0f);
    final_reduce_kernel<<<1, 256, 0, stream>>>(ws, out, NBLOCKS, scale);
}
